// BKNO_71674414235847
// MI455X (gfx1250) — 8-bit matrix operand order repaired and checked by reading, compiled, not run on hardware
//
#include <hip/hip_runtime.h>
#include <math.h>

#define BB   2
#define CH   16
#define TT   64
#define HH   96
#define WW   96
#define NL   4
#define NB   4
#define RHO  2
#define TAPS 27
#define KG   7
#define HB   4
#define TYP  (HB + 2)
#define NPL  (3 * TYP + HB)
#define PLB  (98 * CH)

typedef __attribute__((ext_vector_type(8))) int v8i;
typedef __attribute__((ext_vector_type(4))) int v4i;

__global__ void pack_weights_kernel(const float* __restrict__ logits,
                                    unsigned int* __restrict__ wp)
{
    const int tid = blockIdx.x * blockDim.x + threadIdx.x;
    if (tid >= NL * NB * KG * 32) return;
    const int lane = tid & 31;
    const int g    = (tid >> 5) % KG;
    const int j    = (tid / (32 * KG)) % NB;
    const int l    =  tid / (32 * KG * NB);
    const int m    = lane & 15;
    const int hi   = lane >> 4;

    unsigned int out[8];
#pragma unroll
    for (int v = 0; v < 8; ++v) {
        const int kbase = ((v >= 4) ? 32 : 0) + hi * 16
                        + ((v & 3) * 4);
        unsigned int d = 0u;
#pragma unroll
        for (int by = 0; by < 4; ++by) {
            const int K   = kbase + by;
            const int tgi = K >> 4;
            const int ci  = K & 15;
            const int tap = g * 4 + tgi;
            unsigned int bit = 0u;
            if (tap < TAPS) {
                const float lg =
                    logits[((((size_t)l * NB + j) * CH + m) * CH + ci) * TAPS + tap];
                bit = (lg >= 0.0f) ? 1u : 0u;
            }
            d |= bit << (8 * by);
        }
        out[v] = d;
    }
    unsigned int* dst = wp + ((((size_t)l * NB + j) * KG + g) * 32 + lane) * 8;
#pragma unroll
    for (int v = 0; v < 8; ++v) dst[v] = out[v];
}

__global__ void lift_kernel(const float* __restrict__ m, const float* __restrict__ s,
                            const float* __restrict__ lw, const float* __restrict__ lb,
                            float* __restrict__ X)
{
    size_t idx = (size_t)blockIdx.x * blockDim.x + threadIdx.x;
    const size_t total = (size_t)BB * CH * TT * HH * WW;
    if (idx >= total) return;
    const int w = (int)(idx % WW); size_t r = idx / WW;
    const int h = (int)(r % HH);   r /= HH;
    const int t = (int)(r % TT);   r /= TT;
    const int c = (int)(r % CH);
    const int b = (int)(r / CH);
    const size_t mi = (((size_t)(b * HH + h) * WW) + w) * TT + t;
    X[idx] = lw[c * 2 + 0] * m[mi] + lw[c * 2 + 1] * s[mi] + lb[c];
}

__global__ void proj_kernel(const float* __restrict__ X, const float* __restrict__ pw,
                            const float* __restrict__ pb, float* __restrict__ out)
{
    size_t idx = (size_t)blockIdx.x * blockDim.x + threadIdx.x;
    const size_t total = (size_t)BB * HH * WW * TT;
    if (idx >= total) return;
    const int t = (int)(idx % TT); size_t r = idx / TT;
    const int w = (int)(r % WW);   r /= WW;
    const int h = (int)(r % HH);
    const int b = (int)(r / HH);
    float acc = pb[0];
#pragma unroll
    for (int c = 0; c < CH; ++c)
        acc += pw[c] * X[(((size_t)(b * CH + c) * TT + t) * HH + h) * WW + w];
    out[idx] = acc;
}

__global__ __launch_bounds__(128)
void conv_layer_kernel(const float* __restrict__ X, float* __restrict__ Xn,
                       const unsigned int* __restrict__ wp,
                       const float* __restrict__ thr2,
                       const float* __restrict__ beta_raw2,
                       const float* __restrict__ lamb_raw4,
                       const float* __restrict__ omega1)
{
    __shared__ __align__(16) signed char s_bin[RHO][NPL][PLB];
    __shared__ float s_part[NB][CH][WW];

    const int tid  = threadIdx.x;
    const int wave = tid >> 5;
    const int lane = tid & 31;
    const int lo16 = lane & 15;
    const int hi   = lane >> 4;

    int blk = blockIdx.x;
    const int h0 = (blk % (HH / HB)) * HB; blk /= (HH / HB);
    const int t  = blk % TT;               blk /= TT;
    const int b  = blk;

    const float thr0 = thr2[0];
    const float thr1 = thr2[1];

    for (int idx = tid; idx < NPL * 98 * CH; idx += 128) {
        const int x  = idx % 98;
        const int ci = (idx / 98) % CH;
        const int p  = idx / (98 * CH);
        signed char b0 = 0, b1 = 0;
        if (p < 3 * TYP) {
            const int tt = t  + p / TYP - 1;
            const int hh = h0 + p % TYP - 1;
            const int ww = x - 1;
            if (tt >= 0 && tt < TT && hh >= 0 && hh < HH && ww >= 0 && ww < WW) {
                const float v =
                    X[(((size_t)(b * CH + ci) * TT + tt) * HH + hh) * WW + ww];
                b0 = (v >= thr0) ? 1 : 0;
                b1 = (v >= thr1) ? 1 : 0;
            }
        }
        s_bin[0][p][x * CH + ci] = b0;
        s_bin[1][p][x * CH + ci] = b1;
    }

    int offA[KG], offB[KG];
#pragma unroll
    for (int g = 0; g < KG; ++g) {
        const int ta = g * 4 + hi;
        const int tb = ta + 2;
        offA[g] = ((ta / 9) * TYP + (ta / 3) % 3) * PLB + (ta % 3) * 16;
        offB[g] = ((tb / 9) * TYP + (tb / 3) % 3) * PLB + (tb % 3) * 16;
    }

    v8i afrag[KG];
#pragma unroll
    for (int g = 0; g < KG; ++g) {
        const unsigned int* p = wp + (((size_t)wave * KG + g) * 32 + lane) * 8;
        afrag[g] = *(const v8i*)p;
    }

    const float beta0 = log1pf(expf(beta_raw2[0]));
    const float beta1 = log1pf(expf(beta_raw2[1]));
    const float lamb  = log1pf(expf(lamb_raw4[wave]));
    const float om    = omega1[0];

    __syncthreads();

    for (int r = 0; r < HB; ++r) {
        const int h = h0 + r;
        const size_t rowbase = (((size_t)b * CH * TT + t) * HH + h) * WW;

        if (tid < CH * 3) {
            const int c = tid / 3, seg = tid % 3;
            __builtin_prefetch(
                &X[rowbase + (size_t)c * TT * HH * WW + seg * 32], 0, 3);
        }

        v8i acc[RHO][6];
#pragma unroll
        for (int i = 0; i < RHO; ++i)
#pragma unroll
            for (int n = 0; n < 6; ++n)
                acc[i][n] = (v8i){0, 0, 0, 0, 0, 0, 0, 0};

#pragma unroll
        for (int g = 0; g < KG; ++g) {
#pragma unroll
            for (int i = 0; i < RHO; ++i) {
                const signed char* bi = &s_bin[i][0][0] + r * PLB;
#pragma unroll
                for (int n = 0; n < 6; ++n) {
                    const int colb = (n * 16 + lo16) << 4;
                    const v4i blo = *(const v4i*)(bi + offA[g] + colb);
                    const v4i bhi = *(const v4i*)(bi + offB[g] + colb);
                    v8i bfrag;
                    bfrag[0] = blo[0]; bfrag[1] = blo[1];
                    bfrag[2] = blo[2]; bfrag[3] = blo[3];
                    bfrag[4] = bhi[0]; bfrag[5] = bhi[1];
                    bfrag[6] = bhi[2]; bfrag[7] = bhi[3];
                    acc[i][n] = __builtin_amdgcn_wmma_i32_16x16x64_iu8(
                        false, afrag[g], false, bfrag, acc[i][n], false, false);
                }
            }
        }

#pragma unroll
        for (int n = 0; n < 6; ++n) {
#pragma unroll
            for (int rg = 0; rg < 8; ++rg) {
                const float p = lamb * (beta0 * (float)acc[0][n][rg]
                                      + beta1 * (float)acc[1][n][rg]);
                s_part[wave][rg + 8 * hi][n * 16 + lo16] = p;
            }
        }
        __syncthreads();

        for (int idx = tid; idx < CH * WW; idx += 128) {
            const int c = idx / WW;
            const int w = idx % WW;
            const size_t gi = rowbase + (size_t)c * TT * HH * WW + w;
            float z = om * X[gi]
                    + s_part[0][c][w] + s_part[1][c][w]
                    + s_part[2][c][w] + s_part[3][c][w];
            Xn[gi] = 0.5f * z * (1.0f + erff(z * 0.70710678118654752f));
        }
        __syncthreads();
    }
}

extern "C" void kernel_launch(void* const* d_in, const int* in_sizes, int n_in,
                              void* d_out, int out_size, void* d_ws, size_t ws_size,
                              hipStream_t stream)
{
    (void)in_sizes; (void)n_in; (void)out_size; (void)ws_size;

    const float* m        = (const float*)d_in[0];
    const float* s        = (const float*)d_in[1];
    const float* lift_w   = (const float*)d_in[2];
    const float* lift_b   = (const float*)d_in[3];
    const float* proj_w   = (const float*)d_in[4];
    const float* proj_b   = (const float*)d_in[5];
    const float* thr      = (const float*)d_in[6];
    const float* beta_raw = (const float*)d_in[7];
    const float* klogits  = (const float*)d_in[8];
    const float* lamb_raw = (const float*)d_in[9];
    const float* omega    = (const float*)d_in[10];

    unsigned char* ws = (unsigned char*)d_ws;
    unsigned int* wp  = (unsigned int*)ws;
    const size_t XN   = (size_t)BB * CH * TT * HH * WW;
    float* x0 = (float*)(ws + (1u << 17));
    float* x1 = x0 + XN;

    pack_weights_kernel<<<(NL * NB * KG * 32 + 255) / 256, 256, 0, stream>>>(klogits, wp);

    lift_kernel<<<(int)((XN + 255) / 256), 256, 0, stream>>>(m, s, lift_w, lift_b, x0);

    float* cur = x0;
    float* nxt = x1;
    for (int l = 0; l < NL; ++l) {
        conv_layer_kernel<<<BB * TT * (HH / HB), 128, 0, stream>>>(
            cur, nxt,
            wp + (size_t)l * NB * KG * 32 * 8,
            thr + l * RHO, beta_raw + l * RHO, lamb_raw + l * NB, omega + l);
        float* tmp = cur; cur = nxt; nxt = tmp;
    }

    proj_kernel<<<(BB * HH * WW * TT + 255) / 256, 256, 0, stream>>>(
        cur, proj_w, proj_b, (float*)d_out);
}
